// SelfAttentionPooling_59785944760726
// MI455X (gfx1250) — matrix operand order repaired and checked by reading, not compiled, not run on hardware
//
#include <hip/hip_runtime.h>

// SelfAttentionPooling for MI455X (gfx1250, wave32, WMMA 16x16x32 f16)
// B=32, C=256, H=W=32 -> N=1024, heads=4, head_dim=64.
// qkv GEMM (WMMA, LDS-tiled) -> attention computing S^T=K.Q^T and
// O^T=V^T.P^T so every fragment load is contiguous (A from global K / V^T,
// B from global Q / LDS probs) -> linear pooling folded before out-proj.

typedef _Float16 half_t;
typedef __attribute__((ext_vector_type(16))) _Float16 v16h;
typedef __attribute__((ext_vector_type(8)))  _Float16 h8;
typedef __attribute__((ext_vector_type(4)))  _Float16 h4;
typedef __attribute__((ext_vector_type(8)))  float    v8f;
typedef __attribute__((ext_vector_type(4)))  float    f32x4;

#define NB   32
#define NC   256
#define NHD  4
#define HD   64
#define NPIX 1024

// A fragment: 16x32 f16 from row-major [M][K] source, leading dim ld (halves).
__device__ inline v16h frag_a(const half_t* base, int ld) {
    int lane = threadIdx.x & 31;
    const half_t* p = base + (lane & 15) * ld + ((lane >> 4) << 3);
    v16h f;
    f32x4* q = (f32x4*)&f;
    q[0] = *(const f32x4*)(p);
    q[1] = *(const f32x4*)(p + 16);
    return f;
}

// B fragment: 32x16 f16 (KxN) from TRANSPOSED source [N][K], ld in halves.
__device__ inline v16h frag_b(const half_t* base, int ld) {
    int lane = threadIdx.x & 31;
    const half_t* p = base + (lane & 15) * ld + ((lane >> 4) << 3);
    v16h f;
    f32x4* q = (f32x4*)&f;
    q[0] = *(const f32x4*)(p);
    q[1] = *(const f32x4*)(p + 16);
    return f;
}

// rel-bias separable term: bias(nq,nk) = t(nq) - t(nk)
__device__ inline float rel_t(int n, float p0, float p1) {
    float gy = (float)(n >> 5) * (2.0f / 31.0f) - 1.0f;
    float gx = (float)(n & 31) * (2.0f / 31.0f) - 1.0f;
    return gy * p0 + gx * p1;
}

__global__ void zero_kernel(float* __restrict__ p, int n) {
    int i = blockIdx.x * blockDim.x + threadIdx.x;
    if (i < n) p[i] = 0.0f;
}

// ---- Kernel 1: QKV projection (WMMA GEMM, LDS tiled) --------------------
// grid (N/64, 768/64, B), block 128. Q written pre-scaled by 1/8 into
// [b][h][n][d]; K into [b][h][n][d]; V written TRANSPOSED into [b][h][d][n].
__global__ __launch_bounds__(128) void qkv_kernel(
    const float* __restrict__ x, const float* __restrict__ wqkv,
    const float* __restrict__ bqkv,
    half_t* __restrict__ Qd, half_t* __restrict__ Kd, half_t* __restrict__ Vt) {
    __shared__ half_t smem[64 * 64];     // As(2K) + Bs(2K) halves; reused 64x64
    half_t* As = smem;                   // [m][k] 64x32
    half_t* Bs = smem + 64 * 32;         // [n][k] 64x32 (transposed x tile)
    const int t = threadIdx.x, w = t >> 5, lane = t & 31;
    const int n0 = blockIdx.x * 64, m0 = blockIdx.y * 64, bb = blockIdx.z;

    // destination is uniform per block (m tile of 64 within one of q/k/v)
    const bool isV = (m0 >= 512);
    half_t* dst = (m0 < 256) ? Qd : (m0 < 512 ? Kd : Vt);
    const int cbase = m0 & 255;
    const float sc = (m0 < 256) ? 0.125f : 1.0f;

    v8f acc[4] = {};
    for (int kk = 0; kk < NC; kk += 32) {
        f32x4 ar[4], br[4];
        #pragma unroll
        for (int i = 0; i < 4; ++i) {                  // A: 64x32 as float4
            int lin = t + i * 128;
            int r = lin >> 3, c4 = lin & 7;
            ar[i] = *(const f32x4*)&wqkv[(m0 + r) * NC + kk + c4 * 4];
        }
        #pragma unroll
        for (int i = 0; i < 4; ++i) {                  // B: 32x64 as float4
            int lin = t + i * 128;
            int k = lin >> 4, n4 = lin & 15;
            br[i] = *(const f32x4*)&x[((size_t)bb * NC + kk + k) * NPIX + n0 + n4 * 4];
        }
        __syncthreads();                               // prev compute done
        #pragma unroll
        for (int i = 0; i < 4; ++i) {
            int lin = t + i * 128;
            int r = lin >> 3, c4 = lin & 7;
            h4 hv;
            #pragma unroll
            for (int e = 0; e < 4; ++e) hv[e] = (half_t)ar[i][e];
            *(h4*)&As[r * 32 + c4 * 4] = hv;
        }
        #pragma unroll
        for (int i = 0; i < 4; ++i) {
            int lin = t + i * 128;
            int k = lin >> 4, n4 = lin & 15;
            #pragma unroll
            for (int e = 0; e < 4; ++e) Bs[(n4 * 4 + e) * 32 + k] = (half_t)br[i][e];
        }
        __syncthreads();
        v16h a = frag_a(As + (w * 16) * 32, 32);
        v16h bf[4];
        #pragma unroll
        for (int j = 0; j < 4; ++j) bf[j] = frag_b(Bs + (j * 16) * 32, 32);
        #pragma unroll
        for (int j = 0; j < 4; ++j)
            acc[j] = __builtin_amdgcn_wmma_f32_16x16x32_f16(
                false, a, false, bf[j], (short)0, acc[j], false, false);
    }

    const int mbase = (lane >> 4) * 8, ncol = lane & 15;
    if (!isV) {
        // Q/K: 8 accumulator rows = 8 consecutive d channels -> packed b128
        #pragma unroll
        for (int j = 0; j < 4; ++j) {
            int n = n0 + j * 16 + ncol;
            h8 pv;
            #pragma unroll
            for (int r = 0; r < 8; ++r) {
                int o = m0 + w * 16 + mbase + r;
                pv[r] = (half_t)((acc[j][r] + bqkv[o]) * sc);
            }
            int c = cbase + w * 16 + mbase;
            int h = c >> 6, d0 = c & 63;
            *(h8*)&dst[(((size_t)bb * NHD + h) * NPIX + n) * HD + d0] = pv;
        }
    } else {
        // V: transpose 64x64 tile through LDS, then packed row stores [d][n]
        __syncthreads();
        #pragma unroll
        for (int j = 0; j < 4; ++j)
            #pragma unroll
            for (int r = 0; r < 8; ++r) {
                int o = m0 + w * 16 + mbase + r;
                int dl = w * 16 + mbase + r;           // local channel == d
                smem[dl * 64 + j * 16 + ncol] = (half_t)(acc[j][r] + bqkv[o]);
            }
        __syncthreads();
        const int hgl = cbase >> 6;                    // whole tile is one head
        #pragma unroll
        for (int i = 0; i < 4; ++i) {
            int job = t + i * 128;
            int dl = job >> 3, c8 = job & 7;
            h8 v = *(h8*)&smem[dl * 64 + c8 * 8];
            *(h8*)&dst[(((size_t)bb * NHD + hgl) * HD + dl) * NPIX + n0 + c8 * 8] = v;
        }
    }
}

// ---- Kernel 2: attention core -------------------------------------------
// grid (N/64, B*heads), block 128 (4 waves).
// Phase 1: S^T tiles = K(rows) x Q^T; packed b128 stores into Sl[m][nk] (f16).
// Phase 2: softmax rows of Sl (2 threads/row, 16B vectors).
// Phase 3: O^T tiles = V^T(rows d) x P^T; reduce over m into ymean via shfl.
__global__ __launch_bounds__(128) void attn_kernel(
    const half_t* __restrict__ Q, const half_t* __restrict__ K,
    const half_t* __restrict__ Vt, const float* __restrict__ relp,
    float* __restrict__ ymean) {
    extern __shared__ half_t Sl[];       // [64][1024] f16 = 128 KB
    __shared__ float red[128];

    const int t = threadIdx.x, w = t >> 5, lane = t & 31;
    const int q0 = blockIdx.x * 64;
    const int bh = blockIdx.y, bb = bh >> 2, hh = bh & 3;
    const half_t* Qg  = Q  + (size_t)bh * NPIX * HD;
    const half_t* Kg  = K  + (size_t)bh * NPIX * HD;
    const half_t* Vtg = Vt + (size_t)bh * HD * NPIX;
    const float p0 = relp[hh * 2], p1 = relp[hh * 2 + 1];
    const int mloc = lane & 15, koff = (lane >> 4) * 8;

    // Preload all 8 Q fragments for this 64-row query tile (held in VGPRs)
    v16h qf[4][2];
    #pragma unroll
    for (int jm = 0; jm < 4; ++jm)
        #pragma unroll
        for (int kkh = 0; kkh < 2; ++kkh)
            qf[jm][kkh] = frag_b(Qg + (size_t)(q0 + jm * 16) * HD + kkh * 32, HD);
    float tq[4];
    #pragma unroll
    for (int jm = 0; jm < 4; ++jm) tq[jm] = rel_t(q0 + jm * 16 + mloc, p0, p1);

    // tk advances by a constant when nk += 64 (row index +2): dtk = (4/31)*p0
    float tk[8];
    #pragma unroll
    for (int r = 0; r < 8; ++r) tk[r] = rel_t(w * 16 + koff + r, p0, p1);
    const float dtk = (4.0f / 31.0f) * p0;

    // ---- Phase 1: logits ----
    for (int kb = 0; kb < NPIX; kb += 64) {
        if (kb + 64 < NPIX)   // prefetch next K tile (gfx1250 global_prefetch)
            __builtin_prefetch(Kg + (size_t)(kb + 64) * HD + t * 64, 0, 1);
        v16h ka0 = frag_a(Kg + (size_t)(kb + w * 16) * HD, HD);
        v16h ka1 = frag_a(Kg + (size_t)(kb + w * 16) * HD + 32, HD);
        #pragma unroll
        for (int jm = 0; jm < 4; ++jm) {
            v8f acc = {};
            acc = __builtin_amdgcn_wmma_f32_16x16x32_f16(
                false, ka0, false, qf[jm][0], (short)0, acc, false, false);
            acc = __builtin_amdgcn_wmma_f32_16x16x32_f16(
                false, ka1, false, qf[jm][1], (short)0, acc, false, false);
            h8 pk;
            #pragma unroll
            for (int r = 0; r < 8; ++r)
                pk[r] = (half_t)(acc[r] + tq[jm] - tk[r]);
            *(h8*)&Sl[(size_t)(jm * 16 + mloc) * NPIX + kb + w * 16 + koff] = pk;
        }
        #pragma unroll
        for (int r = 0; r < 8; ++r) tk[r] += dtk;
    }
    __syncthreads();

    // ---- Phase 2: softmax (2 threads per 1024-wide row) ----
    {
        const int row = t >> 1, half = t & 1;
        half_t* rp = Sl + (size_t)row * NPIX + half * 512;
        h8* rp8 = (h8*)rp;
        float m = -1e30f;
        for (int i = 0; i < 64; ++i) {
            h8 v = rp8[i];
            #pragma unroll
            for (int e = 0; e < 8; ++e) m = fmaxf(m, (float)v[e]);
        }
        red[t] = m;
        __syncthreads();
        m = fmaxf(red[row * 2], red[row * 2 + 1]);
        __syncthreads();
        float s = 0.0f;
        for (int i = 0; i < 64; ++i) {
            h8 v = rp8[i], ev;
            #pragma unroll
            for (int e = 0; e < 8; ++e) {
                float ex = __expf((float)v[e] - m);
                s += ex;
                ev[e] = (half_t)ex;
            }
            rp8[i] = ev;
        }
        red[t] = s;
        __syncthreads();
        float inv = 1.0f / (red[row * 2] + red[row * 2 + 1]);
        for (int i = 0; i < 64; ++i) {
            h8 v = rp8[i];
            #pragma unroll
            for (int e = 0; e < 8; ++e) v[e] = (half_t)((float)v[e] * inv);
            rp8[i] = v;
        }
    }
    __syncthreads();

    // ---- Phase 3: O^T = V^T . P^T, accumulate column sums into ymean ----
    v8f oacc[4] = {};
    for (int kb = 0; kb < NPIX; kb += 64) {
        if (kb + 64 < NPIX)
            __builtin_prefetch(Vtg + (size_t)(t >> 1) * NPIX + kb + 64 + (t & 1) * 32, 0, 1);
        #pragma unroll
        for (int kkh = 0; kkh < 2; ++kkh) {
            v16h av = frag_a(Vtg + (size_t)(w * 16) * NPIX + kb + kkh * 32, NPIX);
            #pragma unroll
            for (int j = 0; j < 4; ++j) {
                v16h b = frag_b(Sl + (size_t)(j * 16) * NPIX + kb + kkh * 32, NPIX);
                oacc[j] = __builtin_amdgcn_wmma_f32_16x16x32_f16(
                    false, av, false, b, (short)0, oacc[j], false, false);
            }
        }
    }
    // D[d][m]: lanes within a 16-group hold different m -> shuffle-reduce
    #pragma unroll
    for (int r = 0; r < 8; ++r) {
        float v = oacc[0][r] + oacc[1][r] + oacc[2][r] + oacc[3][r];
        v += __shfl_xor(v, 1);
        v += __shfl_xor(v, 2);
        v += __shfl_xor(v, 4);
        v += __shfl_xor(v, 8);
        if ((lane & 15) == 0) {
            int d = w * 16 + koff + r;
            atomicAdd(&ymean[bb * NC + hh * HD + d], v);
        }
    }
}

// ---- Kernel 3: pooled = w_out . (ymean/N) + b_out -----------------------
__global__ __launch_bounds__(256) void proj_kernel(
    const float* __restrict__ wout, const float* __restrict__ bout,
    const float* __restrict__ ym, float* __restrict__ out) {
    __shared__ float ys[NC];
    int b = blockIdx.x, o = threadIdx.x;
    ys[o] = ym[b * NC + o];
    __syncthreads();
    float s = 0.0f;
    for (int c = 0; c < NC; ++c) s += wout[o * NC + c] * ys[c];
    out[b * NC + o] = s * (1.0f / (float)NPIX) + bout[o];
}

extern "C" void kernel_launch(void* const* d_in, const int* in_sizes, int n_in,
                              void* d_out, int out_size, void* d_ws, size_t ws_size,
                              hipStream_t stream) {
    const float* x    = (const float*)d_in[0];
    const float* wqkv = (const float*)d_in[1];
    const float* bqkv = (const float*)d_in[2];
    const float* wout = (const float*)d_in[3];
    const float* bout = (const float*)d_in[4];
    const float* relp = (const float*)d_in[5];

    const size_t per = (size_t)NB * NHD * NPIX * HD;   // halves per tensor
    half_t* Qws = (half_t*)d_ws;
    half_t* Kws = Qws + per;
    half_t* Vtw = Kws + per;                           // transposed V
    float*  ym  = (float*)(Vtw + per);                 // [B][C] sums over N

    zero_kernel<<<(NB * NC + 255) / 256, 256, 0, stream>>>(ym, NB * NC);

    qkv_kernel<<<dim3(NPIX / 64, 768 / 64, NB), 128, 0, stream>>>(
        x, wqkv, bqkv, Qws, Kws, Vtw);

    size_t smem = (size_t)64 * NPIX * sizeof(half_t);  // 128 KB (2 blocks/WGP)
    (void)hipFuncSetAttribute((const void*)attn_kernel,
                              hipFuncAttributeMaxDynamicSharedMemorySize, (int)smem);
    attn_kernel<<<dim3(NPIX / 64, NB * NHD), 128, smem, stream>>>(
        Qws, Kws, Vtw, relp, ym);

    proj_kernel<<<NB, 256, 0, stream>>>(wout, bout, ym, (float*)d_out);
}
